// SoftKNNPolicy_87660282512066
// MI455X (gfx1250) — compile-verified, matrix operand order checked by reading, not run on hardware
//
#include <hip/hip_runtime.h>

typedef __attribute__((ext_vector_type(16))) __bf16        v16bf;
typedef __attribute__((ext_vector_type(8)))  float         v8f;
typedef __attribute__((ext_vector_type(8)))  unsigned int  v8u;
typedef __bf16 bf16_t;

#define NTILE 512    // N columns per partition workgroup (P = ceil(N/512) WGs)

// ---------- CDNA5 async global->LDS copy (ASYNCcnt-tracked) ----------
__device__ __forceinline__ void async_ld_b128(unsigned ldsoff, const void* gptr) {
  asm volatile("global_load_async_to_lds_b128 %0, %1, off"
               :: "v"(ldsoff), "v"(gptr) : "memory");
}
__device__ __forceinline__ void wait_async0() {
  asm volatile("s_wait_asynccnt 0x0" ::: "memory");
}
// allow the most recent 2 async loads (the in-flight next chunk) to remain outstanding
__device__ __forceinline__ void wait_async2() {
  asm volatile("s_wait_asynccnt 0x2" ::: "memory");
}

// Issue one 64-row x 256-col bf16 chunk (32KB) as per-lane async b128 loads.
// With 1024-thread workgroups each wave contributes exactly 2 ASYNCcnt increments.
__device__ __forceinline__ void issue_chunk(const bf16_t* gsrc, bf16_t* lds_tile) {
  const char* src  = (const char*)gsrc;
  char*       dstg = (char*)lds_tile;
  #pragma unroll
  for (int it = 0; it < 2; ++it) {
    int i = it * 1024 + threadIdx.x;          // 2048 b128 chunks total
    unsigned ldsoff = (unsigned)(unsigned long long)(dstg + (size_t)i * 16);
    async_ld_b128(ldsoff, src + (size_t)i * 16);
  }
}

// ---------- packed f32x2 -> bf16x2 ----------
__device__ __forceinline__ unsigned pk2(float x, float y) {
  __bf16 a = (__bf16)x, b = (__bf16)y;
  unsigned short ua = __builtin_bit_cast(unsigned short, a);
  unsigned short ub = __builtin_bit_cast(unsigned short, b);
  return (unsigned)ua | ((unsigned)ub << 16);
}

// ---------- WMMA fragment load ----------
// 16-bit A/B fragment from a row-major [16 x >=32] tile; rows are the
// M (A) or N (B-from-transposed-source) dimension, K contiguous in memory.
// lane<16 -> row=lane, K-chunks {0..7, 16..23}; lane>=16 -> row=lane-16,
// K-chunks {8..15, 24..31}.
__device__ __forceinline__ v16bf load_frag_bf16(const bf16_t* base, int stride) {
  int lane = threadIdx.x & 31;
  int rc   = lane & 15;
  int ks   = (lane & 16) ? 8 : 0;
  const bf16_t* p = base + rc * stride + ks;
  uint4 c0 = *(const uint4*)(p);
  uint4 c1 = *(const uint4*)(p + 16);
  v8u u;
  u[0] = c0.x; u[1] = c0.y; u[2] = c0.z; u[3] = c0.w;
  u[4] = c1.x; u[5] = c1.y; u[6] = c1.z; u[7] = c1.w;
  return __builtin_bit_cast(v16bf, u);
}

__device__ __forceinline__ v8f wmma_bf16(v16bf a, v16bf b, v8f c) {
  return __builtin_amdgcn_wmma_f32_16x16x32_bf16(false, a, false, b, (short)0, c,
                                                 false, false);
}

static __device__ const v8f kZero8 = {0.f,0.f,0.f,0.f,0.f,0.f,0.f,0.f};

// ---------- prep: Wt[n][k] = bf16(W[k][n]) ----------
__global__ void k_prep_wt(const float* __restrict__ W, bf16_t* __restrict__ Wt) {
  int idx = blockIdx.x * 256 + threadIdx.x;       // 65536 total
  int n = idx >> 8, k = idx & 255;
  Wt[n * 256 + k] = (__bf16)W[k * 256 + n];
}

// ---------- prep: At[a][n] = bf16(Act[n][a]), tiled transpose ----------
__global__ void __launch_bounds__(256) k_act_t(const float* __restrict__ Act,
                                               bf16_t* __restrict__ At, int N) {
  __shared__ float sh[32][129];
  int n0 = blockIdx.x * 32;                        // N % 32 == 0
  #pragma unroll
  for (int it = 0; it < 16; ++it) {
    int o = it * 256 + threadIdx.x;
    int r = o >> 7, a = o & 127;
    sh[r][a] = Act[(long long)(n0 + r) * 128 + a];
  }
  __syncthreads();
  #pragma unroll
  for (int it = 0; it < 16; ++it) {
    int o = it * 256 + threadIdx.x;
    int a = o >> 5, j = o & 31;
    At[(long long)a * N + n0 + j] = (__bf16)sh[j][a];
  }
}

// ---------- encode: Z = bf16( X @ W ) ----------
// X tile (16x256 f32) staged once per workgroup into LDS as bf16; the 4
// column-split waves then read A-fragments via ds_load (no redundant vmem).
__global__ void __launch_bounds__(128) k_encode(const float* __restrict__ X,
                                                const bf16_t* __restrict__ Wt,
                                                bf16_t* __restrict__ Z) {
  __shared__ __align__(16) bf16_t xs[16 * 256];    // 8KB
  int rowb = blockIdx.x * 16;
  int wave = threadIdx.x >> 5;
  int lane = threadIdx.x & 31;
  int nb   = wave * 64;                            // 4 waves cover 256 cols
  // stage + convert: 4096 elems, 8 per thread-iteration
  #pragma unroll
  for (int it = 0; it < 4; ++it) {
    int i = it * 128 + threadIdx.x;                // 512 groups of 8 elems
    int row = i >> 5, c8 = (i & 31) * 8;
    const float* px = X + (long long)(rowb + row) * 256 + c8;
    float4 f0 = *(const float4*)px;
    float4 f1 = *(const float4*)(px + 4);
    uint4 o;
    o.x = pk2(f0.x, f0.y); o.y = pk2(f0.z, f0.w);
    o.z = pk2(f1.x, f1.y); o.w = pk2(f1.z, f1.w);
    *(uint4*)(xs + row * 256 + c8) = o;
  }
  __syncthreads();
  v8f acc[4];
  #pragma unroll
  for (int t = 0; t < 4; ++t) acc[t] = kZero8;
  #pragma unroll
  for (int k = 0; k < 8; ++k) {
    v16bf a = load_frag_bf16(xs + k * 32, 256);
    #pragma unroll
    for (int t = 0; t < 4; ++t) {
      v16bf b = load_frag_bf16(Wt + (nb + t * 16) * 256 + k * 32, 256);
      acc[t] = wmma_bf16(a, b, acc[t]);
    }
  }
  int rhi = (lane & 16) ? 8 : 0;
  #pragma unroll
  for (int t = 0; t < 4; ++t)
    #pragma unroll
    for (int r = 0; r < 8; ++r)
      Z[(long long)(rowb + r + rhi) * 256 + nb + t * 16 + (lane & 15)] = (__bf16)acc[t][r];
}

// ---------- pass 2a: per-partition online row max / exp-sum ----------
// 1024 threads = 32 waves; each wave owns 16 query rows (all 512 queries per
// workgroup), so every z_train slice is streamed exactly once per pass.
// Per-lane online (m,l); one cross-lane merge at the end.
__global__ void __launch_bounds__(1024) k_rowstats(const bf16_t* __restrict__ zq,
                                                   const bf16_t* __restrict__ zt,
                                                   const int* __restrict__ qidx,
                                                   const float* __restrict__ logT,
                                                   float* __restrict__ mpart,
                                                   float* __restrict__ lpart,
                                                   int N, int P) {
  __shared__ __align__(16) bf16_t tile[2][64 * 256];  // 2 x 32KB
  int part = blockIdx.x;
  int wave = threadIdx.x >> 5, lane = threadIdx.x & 31;
  int qbase = wave * 16;
  int rhi = (lane & 16) ? 8 : 0;
  float invT = __expf(-logT[0]);
  int nbeg = part * NTILE;
  int nend = min(nbeg + NTILE, N);
  int nchunks = (nend - nbeg + 63) >> 6;

  int qi[8]; float m[8], l[8];
  #pragma unroll
  for (int r = 0; r < 8; ++r) {
    qi[r] = qidx[qbase + r + rhi];
    m[r] = -3.0e38f; l[r] = 0.f;
  }
  v16bf aq[8];
  #pragma unroll
  for (int k = 0; k < 8; ++k) aq[k] = load_frag_bf16(zq + (long long)qbase * 256 + k * 32, 256);

  issue_chunk(zt + (long long)nbeg * 256, tile[0]);
  for (int c = 0; c < nchunks; ++c) {
    int n0 = nbeg + c * 64;
    int nv = min(64, nend - n0);                   // multiple of 16
    if (c + 1 < nchunks) {
      __syncthreads();                             // all waves done reading buf (c+1)&1
      issue_chunk(zt + (long long)(n0 + 64) * 256, tile[(c + 1) & 1]);
      wait_async2();                               // chunk c landed, c+1 in flight
    } else {
      wait_async0();
    }
    __syncthreads();                               // chunk c visible to all waves
    const bf16_t* buf = tile[c & 1];
    int tiles = nv >> 4;
    for (int t = 0; t < tiles; ++t) {
      v8f s = kZero8;
      #pragma unroll
      for (int k = 0; k < 8; ++k) {
        v16bf b = load_frag_bf16(buf + t * 16 * 256 + k * 32, 256);
        s = wmma_bf16(aq[k], b, s);
      }
      int colg = n0 + t * 16 + (lane & 15);
      #pragma unroll
      for (int r = 0; r < 8; ++r) {
        float x = s[r] * invT;
        if (qi[r] == colg) x = -1.0e30f;
        float mn = fmaxf(m[r], x);
        l[r] = l[r] * __expf(m[r] - mn) + __expf(x - mn);
        m[r] = mn;
      }
    }
  }
  // merge per-lane (m,l) across the 16-lane half
  #pragma unroll
  for (int r = 0; r < 8; ++r) {
    #pragma unroll
    for (int o = 1; o < 16; o <<= 1) {
      float mo = __shfl_xor(m[r], o, 32);
      float lo = __shfl_xor(l[r], o, 32);
      float mn = fmaxf(m[r], mo);
      l[r] = l[r] * __expf(m[r] - mn) + lo * __expf(mo - mn);
      m[r] = mn;
    }
  }
  if ((lane & 15) == 0) {
    #pragma unroll
    for (int r = 0; r < 8; ++r) {
      int row = qbase + r + rhi;
      mpart[(long long)row * P + part] = m[r];
      lpart[(long long)row * P + part] = l[r];
    }
  }
}

// ---------- reduce partials across partitions ----------
__global__ void k_redstats(const float* __restrict__ mpart, const float* __restrict__ lpart,
                           float* __restrict__ mfin, float* __restrict__ linv, int P, int nQ) {
  int row = blockIdx.x * 256 + threadIdx.x;
  if (row >= nQ) return;
  float m = -3.0e38f;
  for (int p = 0; p < P; ++p) m = fmaxf(m, mpart[(long long)row * P + p]);
  float l = 0.f;
  for (int p = 0; p < P; ++p)
    l += lpart[(long long)row * P + p] * __expf(mpart[(long long)row * P + p] - m);
  mfin[row] = m;
  linv[row] = 1.0f / l;
}

__global__ void k_zero(float* __restrict__ p, int n) {
  int i = blockIdx.x * 256 + threadIdx.x;
  if (i < n) p[i] = 0.f;
}

// ---------- pass 2b: weights out + fused pred = P @ V ----------
__global__ void __launch_bounds__(1024) k_output(const bf16_t* __restrict__ zq,
                                                 const bf16_t* __restrict__ zt,
                                                 const bf16_t* __restrict__ At,
                                                 const int* __restrict__ qidx,
                                                 const float* __restrict__ logT,
                                                 const float* __restrict__ mfin,
                                                 const float* __restrict__ linv,
                                                 float* __restrict__ predOut,
                                                 float* __restrict__ wOut,
                                                 int N, int P) {
  __shared__ __align__(16) bf16_t tile[2][64 * 256];  // 2 x 32KB
  __shared__ __align__(16) bf16_t pbuf[32][16 * 64];  // 64KB: per-wave P tile
  int part = blockIdx.x;
  int wave = threadIdx.x >> 5, lane = threadIdx.x & 31;
  int qbase = wave * 16;
  int rhi = (lane & 16) ? 8 : 0;
  float invT = __expf(-logT[0]);
  int nbeg = part * NTILE;
  int nend = min(nbeg + NTILE, N);
  int nchunks = (nend - nbeg + 63) >> 6;

  int qi[8]; float mr[8], lr[8];
  #pragma unroll
  for (int r = 0; r < 8; ++r) {
    int row = qbase + r + rhi;
    qi[r] = qidx[row]; mr[r] = mfin[row]; lr[r] = linv[row];
  }
  v16bf aq[8];
  #pragma unroll
  for (int k = 0; k < 8; ++k) aq[k] = load_frag_bf16(zq + (long long)qbase * 256 + k * 32, 256);

  v8f accP[8];
  #pragma unroll
  for (int j = 0; j < 8; ++j) accP[j] = kZero8;

  issue_chunk(zt + (long long)nbeg * 256, tile[0]);
  for (int c = 0; c < nchunks; ++c) {
    int n0 = nbeg + c * 64;
    int nv = min(64, nend - n0);                   // multiple of 16
    if (c + 1 < nchunks) {
      __syncthreads();
      issue_chunk(zt + (long long)(n0 + 64) * 256, tile[(c + 1) & 1]);
      wait_async2();
    } else {
      wait_async0();
    }
    __syncthreads();
    const bf16_t* buf = tile[c & 1];
    int tiles = nv >> 4;
    for (int t = 0; t < tiles; ++t) {
      v8f s = kZero8;
      #pragma unroll
      for (int k = 0; k < 8; ++k) {
        v16bf b = load_frag_bf16(buf + t * 16 * 256 + k * 32, 256);
        s = wmma_bf16(aq[k], b, s);
      }
      int colL = t * 16 + (lane & 15);
      int colg = n0 + colL;
      #pragma unroll
      for (int r = 0; r < 8; ++r) {
        float x = s[r] * invT;
        if (qi[r] == colg) x = -1.0e30f;
        float w = __expf(x - mr[r]) * lr[r];       // normalized weight
        wOut[(long long)(qbase + r + rhi) * N + colg] = w;
        pbuf[wave][(r + rhi) * 64 + colL] = (__bf16)w;
      }
    }
    // fused pred accumulation: P(16 x nv) @ At-slice(nv x 128)
    int ksteps = nv >> 5;                          // nv=64 -> 2, nv=32 -> 1
    for (int ks = 0; ks < ksteps; ++ks) {
      v16bf pa = load_frag_bf16(&pbuf[wave][0] + ks * 32, 64);
      #pragma unroll
      for (int j = 0; j < 8; ++j) {
        v16bf bv = load_frag_bf16(At + (long long)(j * 16) * N + n0 + ks * 32, N);
        accP[j] = wmma_bf16(pa, bv, accP[j]);
      }
    }
  }
  #pragma unroll
  for (int j = 0; j < 8; ++j)
    #pragma unroll
    for (int r = 0; r < 8; ++r)
      atomicAdd(&predOut[(long long)(qbase + r + rhi) * 128 + j * 16 + (lane & 15)],
                accP[j][r]);
}

extern "C" void kernel_launch(void* const* d_in, const int* in_sizes, int n_in,
                              void* d_out, int out_size, void* d_ws, size_t ws_size,
                              hipStream_t stream) {
  (void)n_in; (void)out_size; (void)ws_size;
  const float* query_obs = (const float*)d_in[0];
  const int*   qidx      = (const int*)d_in[1];
  const float* W_enc     = (const float*)d_in[2];
  const float* train_obs = (const float*)d_in[3];
  const float* train_act = (const float*)d_in[4];
  const float* logT      = (const float*)d_in[5];

  const int nQ = in_sizes[0] / 256;     // 512
  const int N  = in_sizes[3] / 256;     // 100000
  const int P  = (N + NTILE - 1) / NTILE;

  float* pred = (float*)d_out;                       // [nQ x 128]
  float* wOut = pred + (size_t)nQ * 128;             // [nQ x N]

  // workspace carve-up (all 16B aligned); mpart/lpart after zt also serve as
  // slack for the (never-read) tail over-fetch of the last async chunk.
  char* ws = (char*)d_ws;
  size_t off = 0;
  bf16_t* Wt = (bf16_t*)(ws + off); off += (size_t)256 * 256 * 2;
  bf16_t* At = (bf16_t*)(ws + off); off += (size_t)128 * N * 2;
  bf16_t* zq = (bf16_t*)(ws + off); off += (size_t)nQ * 256 * 2;
  bf16_t* zt = (bf16_t*)(ws + off); off += (size_t)N * 256 * 2;
  float* mpart = (float*)(ws + off); off += (((size_t)nQ * P * 4) + 15) & ~(size_t)15;
  float* lpart = (float*)(ws + off); off += (((size_t)nQ * P * 4) + 15) & ~(size_t)15;
  float* mfin  = (float*)(ws + off); off += (((size_t)nQ * 4) + 15) & ~(size_t)15;
  float* linv  = (float*)(ws + off);

  // 1) transposed bf16 weights + actions
  k_prep_wt<<<256, 256, 0, stream>>>(W_enc, Wt);
  k_act_t<<<N / 32, 256, 0, stream>>>(train_act, At, N);

  // 2) encode query + train observations
  k_encode<<<nQ / 16, 128, 0, stream>>>(query_obs, Wt, zq);
  k_encode<<<N  / 16, 128, 0, stream>>>(train_obs, Wt, zt);

  // 3) row stats (online max / exp-sum per N-partition), then reduce
  k_rowstats<<<P, 1024, 0, stream>>>(zq, zt, qidx, logT, mpart, lpart, N, P);
  k_redstats<<<(nQ + 255) / 256, 256, 0, stream>>>(mpart, lpart, mfin, linv, P, nQ);

  // 4) zero pred, then fused weights-out + pred accumulation
  k_zero<<<(nQ * 128 + 255) / 256, 256, 0, stream>>>(pred, nQ * 128);
  k_output<<<P, 1024, 0, stream>>>(zq, zt, At, qidx, logT, mfin, linv,
                                   pred, wOut, N, P);
}
